// GeodesicAttention_83365315215717
// MI455X (gfx1250) — matrix operand order repaired and checked by reading, not compiled, not run on hardware
//
#include <hip/hip_runtime.h>

typedef __attribute__((ext_vector_type(16))) __bf16 v16bf;
typedef __attribute__((ext_vector_type(8)))  __bf16 v8bf;
typedef __attribute__((ext_vector_type(4)))  __bf16 v4bf;
typedef __attribute__((ext_vector_type(8)))  float  v8f;

#define GA_B 8
#define GA_L 4096
#define GA_D 128
#define GA_EPS 1e-8f

// Fast-path LDS: strip 16x4096 fp32 + rowsum[16] + sqRow[16]
#define GA_LDS_FAST  (16 * GA_L * 4 + 128)
// Fallback LDS: strip + Ahi/Alo (16x128 bf16 each) + rowsum + sqRow
#define GA_LDS_FALL  (16 * GA_L * 4 + 2 * 16 * GA_D * 2 + 128)

union GaV16 { v16bf v; v8bf h[2]; };

// ---------------------------------------------------------------------------
// Prep kernel (fast path): one wave32 per row; computes sq[row] AND the
// bf16 hi/lo split of x, written row-major so B fragments are contiguous.
// ---------------------------------------------------------------------------
__global__ void ga_prep_kernel(const float* __restrict__ x,
                               float* __restrict__ sqg,
                               __bf16* __restrict__ xhi,
                               __bf16* __restrict__ xlo) {
    const int wave = threadIdx.x >> 5;
    const int lane = threadIdx.x & 31;
    const int row  = blockIdx.x * 8 + wave;                 // [0, B*L)
    const float4 f = ((const float4*)(x + (size_t)row * GA_D))[lane];
    const float fv[4] = { f.x, f.y, f.z, f.w };

    float s = f.x * f.x + f.y * f.y + f.z * f.z + f.w * f.w;
    #pragma unroll
    for (int off = 16; off > 0; off >>= 1) s += __shfl_xor(s, off, 32);
    if (lane == 0) sqg[row] = s;

    v4bf hv, lv;
    #pragma unroll
    for (int i = 0; i < 4; ++i) {
        const __bf16 h = (__bf16)fv[i];
        hv[i] = h;
        lv[i] = (__bf16)(fv[i] - (float)h);
    }
    *(v4bf*)&xhi[(size_t)row * GA_D + lane * 4] = hv;
    *(v4bf*)&xlo[(size_t)row * GA_D + lane * 4] = lv;
}

// ---------------------------------------------------------------------------
// Shared epilogue: dist -> exp -> strip/rowsum, then normalize + stream out.
// ---------------------------------------------------------------------------
__device__ __forceinline__ void ga_tile_epilogue(
    const v8f& acc0, const v8f& acc1, const v8f& acc2,
    const float* sr, float sqc, int l0, int col, int laneHi,
    float* strip, float* racc) {
    #pragma unroll
    for (int v = 0; v < 8; ++v) {
        const int M = v + 8 * laneHi;
        const float inner = acc0[v] + acc1[v] + acc2[v];
        float dsq = sr[v] + sqc - 2.0f * inner;
        if (l0 + M == col) dsq = 0.0f;                    // exact diagonal
        const float dist = __builtin_sqrtf(fmaxf(dsq, 0.0f) + GA_EPS);
        const float e = __expf(-dist);                    // exp<=1: no max pass
        strip[M * GA_L + col] = e;
        racc[v] += e;
    }
}

__device__ __forceinline__ void ga_normalize_store(
    float* strip, float* rowsum, float* racc, int tid, int laneHi,
    float* attn, int b, int l0) {
    #pragma unroll
    for (int v = 0; v < 8; ++v) atomicAdd(&rowsum[v + 8 * laneHi], racc[v]);
    __syncthreads();
    if (tid < 16) rowsum[tid] = 1.0f / rowsum[tid];
    __syncthreads();

    float* outp = attn + ((size_t)b * GA_L + l0) * (size_t)GA_L;
    #pragma unroll 4
    for (int i = 0; i < 64; ++i) {
        const int idx4 = tid + i * 256;                   // float4 index 0..16383
        const float inv = rowsum[idx4 >> 10];
        float4 val = *(float4*)&strip[idx4 * 4];
        val.x *= inv; val.y *= inv; val.z *= inv; val.w *= inv;
        *(float4*)(outp + (size_t)idx4 * 4) = val;
    }
}

// ---------------------------------------------------------------------------
// Fast Gram kernel: pre-split bf16 inputs; A strip fragments held in VGPRs
// for the whole strip; inner loop = 8 vector loads + 12 WMMAs + epilogue.
// ---------------------------------------------------------------------------
__global__ void __launch_bounds__(256)
ga_gram_pre_kernel(const __bf16* __restrict__ xhi,
                   const __bf16* __restrict__ xlo,
                   const float* __restrict__ sqg,
                   float* __restrict__ attn) {
    extern __shared__ char smem[];
    float* strip  = (float*)smem;                         // 16*4096
    float* rowsum = (float*)(smem + 16 * GA_L * 4);       // 16
    float* sqRow  = rowsum + 16;                          // 16

    const int tid = threadIdx.x;
    const int b   = blockIdx.y;
    const int l0  = blockIdx.x * 16;
    const float* sqb = sqg + (size_t)b * GA_L;

    if (tid < 16) { rowsum[tid] = 0.0f; sqRow[tid] = sqb[l0 + tid]; }
    __syncthreads();

    const int wave   = tid >> 5;
    const int lane   = tid & 31;
    const int laneHi = lane >> 4;
    const int N      = lane & 15;

    float sr[8];
    #pragma unroll
    for (int v = 0; v < 8; ++v) sr[v] = sqRow[v + 8 * laneHi];

    // A fragments for all 4 K-chunks, hi and lo, kept in registers.
    // Layout (16-bit A 16x32): lane M = lane&15;
    // elems 0-15 -> K = kb + laneHi*16 + e.
    const __bf16* arh = xhi + (size_t)(b * GA_L + l0 + N) * GA_D;
    const __bf16* arl = xlo + (size_t)(b * GA_L + l0 + N) * GA_D;
    GaV16 ah[4], al[4];
    #pragma unroll
    for (int kc = 0; kc < 4; ++kc) {
        const int k0 = kc * 32 + laneHi * 16;
        ah[kc].h[0] = *(const v8bf*)(arh + k0);
        ah[kc].h[1] = *(const v8bf*)(arh + k0 + 8);
        al[kc].h[0] = *(const v8bf*)(arl + k0);
        al[kc].h[1] = *(const v8bf*)(arl + k0 + 8);
    }

    const __bf16* xbh = xhi + (size_t)b * GA_L * GA_D;
    const __bf16* xbl = xlo + (size_t)b * GA_L * GA_D;

    float racc[8];
    #pragma unroll
    for (int v = 0; v < 8; ++v) racc[v] = 0.0f;

    for (int t = 0; t < 32; ++t) {
        const int col = wave * 512 + t * 16 + N;
        const float sqc = sqb[col];
        // B fragment (32x16): lane N = lane&15, elem e -> K = kb + 16*laneHi + e
        const __bf16* bh = xbh + (size_t)col * GA_D + laneHi * 16;
        const __bf16* bl = xbl + (size_t)col * GA_D + laneHi * 16;

        v8f acc0 = {}; v8f acc1 = {}; v8f acc2 = {};
        #pragma unroll
        for (int kc = 0; kc < 4; ++kc) {
            const v16bf bhv = *(const v16bf*)(bh + kc * 32);
            const v16bf blv = *(const v16bf*)(bl + kc * 32);
            acc0 = __builtin_amdgcn_wmma_f32_16x16x32_bf16(
                       false, ah[kc].v, false, bhv, (short)0, acc0, false, false);
            acc1 = __builtin_amdgcn_wmma_f32_16x16x32_bf16(
                       false, ah[kc].v, false, blv, (short)0, acc1, false, false);
            acc2 = __builtin_amdgcn_wmma_f32_16x16x32_bf16(
                       false, al[kc].v, false, bhv, (short)0, acc2, false, false);
        }
        ga_tile_epilogue(acc0, acc1, acc2, sr, sqc, l0, col, laneHi, strip, racc);
    }
    ga_normalize_store(strip, rowsum, racc, tid, laneHi, attn, b, l0);
}

// ---------------------------------------------------------------------------
// Fallback sq kernel + Gram kernel (on-the-fly bf16 split), used only if the
// workspace cannot hold the pre-split arrays.
// ---------------------------------------------------------------------------
__global__ void ga_sq_kernel(const float* __restrict__ x, float* __restrict__ sqg) {
    const int wave = threadIdx.x >> 5;
    const int lane = threadIdx.x & 31;
    const int row  = blockIdx.x * 8 + wave;
    const float4 f = ((const float4*)(x + (size_t)row * GA_D))[lane];
    float s = f.x * f.x + f.y * f.y + f.z * f.z + f.w * f.w;
    #pragma unroll
    for (int off = 16; off > 0; off >>= 1) s += __shfl_xor(s, off, 32);
    if (lane == 0) sqg[row] = s;
}

__global__ void __launch_bounds__(256)
ga_gram_fallback_kernel(const float* __restrict__ x,
                        const float* __restrict__ sqg,
                        float* __restrict__ attn) {
    extern __shared__ char smem[];
    float*  strip  = (float*)smem;
    __bf16* Ahi    = (__bf16*)(smem + 16 * GA_L * 4);
    __bf16* Alo    = Ahi + 16 * GA_D;
    float*  rowsum = (float*)(Alo + 16 * GA_D);
    float*  sqRow  = rowsum + 16;

    const int tid = threadIdx.x;
    const int b   = blockIdx.y;
    const int l0  = blockIdx.x * 16;
    const float* xb  = x   + (size_t)b * GA_L * GA_D;
    const float* sqb = sqg + (size_t)b * GA_L;

    for (int t = tid; t < 16 * GA_D; t += 256) {
        const int r = t >> 7, k = t & 127;
        const float f = xb[(size_t)(l0 + r) * GA_D + k];
        const __bf16 h = (__bf16)f;
        Ahi[r * GA_D + k] = h;
        Alo[r * GA_D + k] = (__bf16)(f - (float)h);
    }
    if (tid < 16) { rowsum[tid] = 0.0f; sqRow[tid] = sqb[l0 + tid]; }
    __syncthreads();

    const int wave   = tid >> 5;
    const int lane   = tid & 31;
    const int laneHi = lane >> 4;
    const int N      = lane & 15;

    float sr[8];
    #pragma unroll
    for (int v = 0; v < 8; ++v) sr[v] = sqRow[v + 8 * laneHi];
    float racc[8];
    #pragma unroll
    for (int v = 0; v < 8; ++v) racc[v] = 0.0f;

    for (int t = 0; t < 32; ++t) {
        const int col = wave * 512 + t * 16 + N;
        const float sqc = sqb[col];
        v8f acc0 = {}; v8f acc1 = {}; v8f acc2 = {};
        #pragma unroll
        for (int kc = 0; kc < 4; ++kc) {
            const int kb = kc * 32;
            GaV16 ah, al;
            const int M  = N;
            const int k0 = kb + laneHi * 16;
            ah.h[0] = *(const v8bf*)&Ahi[M * GA_D + k0];
            ah.h[1] = *(const v8bf*)&Ahi[M * GA_D + k0 + 8];
            al.h[0] = *(const v8bf*)&Alo[M * GA_D + k0];
            al.h[1] = *(const v8bf*)&Alo[M * GA_D + k0 + 8];

            const float4* bp4 =
                (const float4*)(xb + (size_t)col * GA_D + kb + laneHi * 16);
            const float4 f0 = bp4[0], f1 = bp4[1], f2 = bp4[2], f3 = bp4[3];
            const float bfv[16] = { f0.x, f0.y, f0.z, f0.w,
                                    f1.x, f1.y, f1.z, f1.w,
                                    f2.x, f2.y, f2.z, f2.w,
                                    f3.x, f3.y, f3.z, f3.w };
            v16bf bhi, blo;
            #pragma unroll
            for (int e = 0; e < 16; ++e) {
                const __bf16 h = (__bf16)bfv[e];
                bhi[e] = h;
                blo[e] = (__bf16)(bfv[e] - (float)h);
            }
            acc0 = __builtin_amdgcn_wmma_f32_16x16x32_bf16(
                       false, ah.v, false, bhi, (short)0, acc0, false, false);
            acc1 = __builtin_amdgcn_wmma_f32_16x16x32_bf16(
                       false, ah.v, false, blo, (short)0, acc1, false, false);
            acc2 = __builtin_amdgcn_wmma_f32_16x16x32_bf16(
                       false, al.v, false, bhi, (short)0, acc2, false, false);
        }
        ga_tile_epilogue(acc0, acc1, acc2, sr, sqc, l0, col, laneHi, strip, racc);
    }
    ga_normalize_store(strip, rowsum, racc, tid, laneHi, attn, b, l0);
}

// ---------------------------------------------------------------------------
// Kernel: x_agg / v_agg from the last attn row (tiny; L2-resident reads).
// ---------------------------------------------------------------------------
__global__ void ga_agg_kernel(const float* __restrict__ x,
                              const float* __restrict__ v,
                              const float* __restrict__ attn,
                              float* __restrict__ out) {
    const int b = blockIdx.x;
    const int d = threadIdx.x;                            // 0..127
    const float* w  = attn + ((size_t)b * GA_L + (GA_L - 1)) * (size_t)GA_L;
    const float* xb = x + (size_t)b * GA_L * GA_D + d;
    const float* vb = v + (size_t)b * GA_L * GA_D + d;
    float ax = 0.0f, av = 0.0f;
    for (int m = 0; m < GA_L; ++m) {
        const float ww = w[m];
        ax = fmaf(ww, xb[(size_t)m * GA_D], ax);
        av = fmaf(ww, vb[(size_t)m * GA_D], av);
    }
    out[b * GA_D + d]               = ax;                 // x_agg
    out[GA_B * GA_D + b * GA_D + d] = av;                 // v_agg
}

// ---------------------------------------------------------------------------
extern "C" void kernel_launch(void* const* d_in, const int* in_sizes, int n_in,
                              void* d_out, int out_size, void* d_ws, size_t ws_size,
                              hipStream_t stream) {
    (void)in_sizes; (void)n_in; (void)out_size;
    const float* x = (const float*)d_in[0];
    const float* v = (const float*)d_in[1];
    float* out  = (float*)d_out;
    float* attn = out + 2 * GA_B * GA_D;

    const size_t nSq  = (size_t)GA_B * GA_L;              // fp32
    const size_t nEl  = (size_t)GA_B * GA_L * GA_D;       // bf16 each for hi/lo
    const size_t need = nSq * 4 + 2 * nEl * 2;            // 128KB + 16MB

    float* sqg = (float*)d_ws;

    if (ws_size >= need) {
        __bf16* xhi = (__bf16*)((char*)d_ws + nSq * 4);
        __bf16* xlo = xhi + nEl;
        ga_prep_kernel<<<GA_B * GA_L / 8, 256, 0, stream>>>(x, sqg, xhi, xlo);
        ga_gram_pre_kernel<<<dim3(GA_L / 16, GA_B), 256, GA_LDS_FAST, stream>>>(
            xhi, xlo, sqg, attn);
    } else {
        ga_sq_kernel<<<GA_B * GA_L / 8, 256, 0, stream>>>(x, sqg);
        ga_gram_fallback_kernel<<<dim3(GA_L / 16, GA_B), 256, GA_LDS_FALL, stream>>>(
            x, sqg, attn);
    }

    ga_agg_kernel<<<GA_B, GA_D, 0, stream>>>(x, v, attn, out);
}
